// Inception1_23287312679187
// MI455X (gfx1250) — matrix operand order repaired and checked by reading, not compiled, not run on hardware
//
#include <hip/hip_runtime.h>
#include <hip/hip_bf16.h>
#include <math.h>

// ---------------- types ----------------
typedef __attribute__((ext_vector_type(16))) __bf16 v16bf;
typedef __attribute__((ext_vector_type(8)))  __bf16 v8bf;
typedef __attribute__((ext_vector_type(8)))  float  v8f;

#define NB    256      // batch
#define NH    15       // rows
#define WFULL 2400
#define TT    240      // timesteps
#define NPAIR 105
#define FEAT  375
#define KPAD  384
#define HID   30
#define G3    90       // 3*HID
#define GPAD  96
#define MROWS (TT*NB)  // 61440

// ---------------- WMMA helpers ----------------
__device__ inline v8f wmma_bf16(v16bf a, v16bf b, v8f c) {
  return __builtin_amdgcn_wmma_f32_16x16x32_bf16(false, a, false, b, (short)0, c,
                                                 false, false);
}

// A-matrix 16x32 bf16 fragment: lane L (<16): M=L, K = {0..7,16..23}; L>=16: M=L-16, K={8..15,24..31}
__device__ inline v16bf load_a_frag(const __bf16* tile_base, int ld, int k0, int lane) {
  int m = lane & 15;
  const __bf16* p = tile_base + (size_t)m * ld + k0 + ((lane < 16) ? 0 : 8);
  v8bf lo = *(const v8bf*)p;          // k offsets +0..7
  v8bf hi = *(const v8bf*)(p + 16);   // k offsets +16..23
  v16bf r;
#pragma unroll
  for (int e = 0; e < 8; ++e) { r[e] = lo[e]; r[8 + e] = hi[e]; }
  return r;
}

// B-matrix 32x16 bf16 fragment from row-major W[g][k]: lane col = lane%16, K = (lane<16?{0..7,16..23}:{8..15,24..31})+k0
__device__ inline v16bf load_b_frag(const __bf16* W, int g, int ld, int k0, int lane) {
  const __bf16* p = W + (size_t)g * ld + k0 + ((lane < 16) ? 0 : 8);
  v8bf lo = *(const v8bf*)p;
  v8bf hi = *(const v8bf*)(p + 16);
  v16bf r;
#pragma unroll
  for (int e = 0; e < 8; ++e) { r[e] = lo[e]; r[8 + e] = hi[e]; }
  return r;
}

// ---------------- kernel 0: zero stats ----------------
__global__ void k_init(float* stats) {
  if (threadIdx.x < 14) stats[threadIdx.x] = 0.0f;
}

// ---------------- kernel 1: features + grouped stats ----------------
__global__ __launch_bounds__(256) void k_features(const float* __restrict__ x,
                                                  __bf16* __restrict__ feat,
                                                  float* __restrict__ stats) {
  const int t = blockIdx.x;        // 0..239
  const int b = threadIdx.x;       // 0..255
  const size_t m = (size_t)t * NB + b;
  const float* p = x + (size_t)b * (NH * WFULL) + t * 10;
  __bf16* fr = feat + m * KPAD;

  float sp[NH][10];   // spread
  float mn[NH], sd[NH];
  float part[14];
#pragma unroll
  for (int q = 0; q < 14; ++q) part[q] = 0.0f;

#pragma unroll
  for (int h = 0; h < NH; ++h) {
    float v[10]; float s = 0.0f;
#pragma unroll
    for (int w = 0; w < 10; ++w) { v[w] = p[h * WFULL + w]; s += v[w]; }
    float mu = s * 0.1f;
    float ss = 0.0f, dec = 0.0f;
#pragma unroll
    for (int w = 0; w < 10; ++w) {
      float d = v[w] - mu;
      sp[h][w] = d;
      ss += d * d;
      dec += d * ((float)(w + 1) * (1.0f / 55.0f));
    }
    float st = sqrtf(ss * (1.0f / 9.0f));
    float decay = dec + mu;                    // sum(xw*dw), sum(dw)=1
    float zs = mu / (st + 0.01f);
    mn[h] = mu; sd[h] = st;
    fr[210 + h] = (__bf16)decay;
    fr[225 + h] = (__bf16)st;
    fr[240 + h] = (__bf16)mu;
    fr[255 + h] = (__bf16)zs;
    part[2] += decay; part[9]  += decay * decay;
    part[3] += st;    part[10] += st * st;
    part[4] += mu;    part[11] += mu * mu;
    part[5] += zs;    part[12] += zs * zs;
  }

  int pi = 0;
#pragma unroll
  for (int a = 0; a < NH - 1; ++a) {
#pragma unroll
    for (int c = a + 1; c < NH; ++c) {
      float cv = 0.0f;
#pragma unroll
      for (int w = 0; w < 10; ++w) cv += sp[a][w] * sp[c][w];
      cv *= (1.0f / 9.0f);
      float cr = cv / (sd[a] * sd[c] + 0.01f) * 0.9f;
      float ml = 0.9f * cv + mn[a] * mn[c];    // sum(xa*xb)/10 identity
      fr[pi]         = (__bf16)cv;
      fr[105 + pi]   = (__bf16)cr;
      fr[270 + pi]   = (__bf16)ml;
      part[0] += cv; part[7]  += cv * cv;
      part[1] += cr; part[8]  += cr * cr;
      part[6] += ml; part[13] += ml * ml;
      ++pi;
    }
  }
#pragma unroll
  for (int k = FEAT; k < KPAD; ++k) fr[k] = (__bf16)0.0f;

  // block reduction of 14 partials, then global atomic add
  __shared__ float red[256];
  // note: part order => stats[q] for q<7 sums, q>=7 sumsq; remap to (sum[g],sq[g]) pairs
  float sums[14];
  sums[0] = part[0]; sums[7]  = part[7];   // cov
  sums[1] = part[1]; sums[8]  = part[8];   // corr
  sums[2] = part[2]; sums[9]  = part[9];   // decay
  sums[3] = part[3]; sums[10] = part[10];  // std
  sums[4] = part[4]; sums[11] = part[11];  // mean
  sums[5] = part[5]; sums[12] = part[12];  // zscore
  sums[6] = part[6]; sums[13] = part[13];  // mul
  for (int q = 0; q < 14; ++q) {
    red[threadIdx.x] = sums[q];
    __syncthreads();
    for (int off = 128; off > 0; off >>= 1) {
      if (threadIdx.x < off) red[threadIdx.x] += red[threadIdx.x + off];
      __syncthreads();
    }
    if (threadIdx.x == 0) atomicAdd(&stats[q], red[0]);
    __syncthreads();
  }
}

// ---------------- kernel 2: finalize BN scale/shift per group ----------------
__global__ void k_finalize(const float* __restrict__ stats,
                           const float* __restrict__ bnp,
                           float* __restrict__ ac) {
  int g = threadIdx.x;
  if (g < 7) {
    const float cnt[7] = {105.f, 105.f, 15.f, 15.f, 15.f, 15.f, 105.f};
    float n = cnt[g] * (float)MROWS;
    float mu = stats[g] / n;
    float var = stats[7 + g] / n - mu * mu;
    float a = bnp[2 * g] * rsqrtf(var + 1e-5f);
    float c = bnp[2 * g + 1] - mu * a;
    ac[g] = a;
    ac[7 + g] = c;
  }
}

__device__ inline int grp_of(int k) {
  return k < 105 ? 0 : k < 210 ? 1 : k < 225 ? 2 : k < 240 ? 3 : k < 255 ? 4
                                                                : k < 270 ? 5 : 6;
}

// ---------------- kernel 3: BN-folded, padded bf16 weights ----------------
__global__ __launch_bounds__(128) void k_wprep(
    const float* __restrict__ Wih0, const float* __restrict__ bih0,
    const float* __restrict__ Whh0, const float* __restrict__ bhh0,
    const float* __restrict__ Wih1, const float* __restrict__ bih1,
    const float* __restrict__ Whh1, const float* __restrict__ bhh1,
    const float* __restrict__ ac,
    __bf16* __restrict__ wp0, float* __restrict__ b0p,
    __bf16* __restrict__ whh0p, float* __restrict__ bhh0p,
    __bf16* __restrict__ wih1p, float* __restrict__ bih1p,
    __bf16* __restrict__ whh1p, float* __restrict__ bhh1p) {
  int g = threadIdx.x;
  if (g >= GPAD) return;
  float av[7], cv[7];
#pragma unroll
  for (int i = 0; i < 7; ++i) { av[i] = ac[i]; cv[i] = ac[7 + i]; }
  float bias = 0.0f;
  for (int k = 0; k < KPAD; ++k) {
    float w = (g < G3 && k < FEAT) ? Wih0[g * FEAT + k] : 0.0f;
    int gr = (k < FEAT) ? grp_of(k) : 0;
    wp0[(size_t)g * KPAD + k] = (__bf16)(w * av[gr]);
    bias += w * cv[gr];
  }
  b0p[g] = bias + (g < G3 ? bih0[g] : 0.0f);
  for (int k = 0; k < 32; ++k) {
    bool ok = (g < G3 && k < HID);
    whh0p[g * 32 + k] = (__bf16)(ok ? Whh0[g * HID + k] : 0.0f);
    wih1p[g * 32 + k] = (__bf16)(ok ? Wih1[g * HID + k] : 0.0f);
    whh1p[g * 32 + k] = (__bf16)(ok ? Whh1[g * HID + k] : 0.0f);
  }
  bhh0p[g] = (g < G3) ? bhh0[g] : 0.0f;
  bih1p[g] = (g < G3) ? bih1[g] : 0.0f;
  bhh1p[g] = (g < G3) ? bhh1[g] : 0.0f;
}

// ---------------- kernel 4: big WMMA GEMM  xp0[m][g] = feat[m][:] . wp0[g][:] + b0p[g] ----------------
__global__ __launch_bounds__(192) void k_gemm(const __bf16* __restrict__ feat,
                                              const __bf16* __restrict__ wp0,
                                              const float* __restrict__ b0p,
                                              float* __restrict__ xp0) {
  const int lane = threadIdx.x & 31;
  const int ntile = threadIdx.x >> 5;   // 0..5
  const int mtile = blockIdx.x;         // 0..3839
  const int g = ntile * 16 + (lane & 15);
  v8f acc;
  float bias = b0p[g];
#pragma unroll
  for (int r = 0; r < 8; ++r) acc[r] = bias;

  const __bf16* atile = feat + (size_t)mtile * 16 * KPAD;
#pragma unroll
  for (int kt = 0; kt < KPAD / 32; ++kt) {
    v16bf a = load_a_frag(atile, KPAD, kt * 32, lane);
    v16bf b = load_b_frag(wp0, g, KPAD, kt * 32, lane);
    acc = wmma_bf16(a, b, acc);
  }
  const int rowbase = mtile * 16 + ((lane < 16) ? 0 : 8);
#pragma unroll
  for (int r = 0; r < 8; ++r) xp0[(size_t)(rowbase + r) * GPAD + g] = acc[r];
}

// ---------------- kernel 5: fused 2-layer GRU scan (16 batch rows / block) ----------------
__global__ __launch_bounds__(192) void k_scan(const float* __restrict__ xp0,
                                              const __bf16* __restrict__ whh0p,
                                              const float* __restrict__ bhh0p,
                                              const __bf16* __restrict__ wih1p,
                                              const float* __restrict__ bih1p,
                                              const __bf16* __restrict__ whh1p,
                                              const float* __restrict__ bhh1p,
                                              float* __restrict__ h1out) {
  __shared__ __bf16 h0[16][32];
  __shared__ __bf16 h1[16][32];
  __shared__ float h0f[16][HID];
  __shared__ float h1f[16][HID];
  __shared__ float hpA[16][GPAD];
  __shared__ float hpB[16][GPAD];

  const int tid = threadIdx.x;
  const int lane = tid & 31;
  const int ntile = tid >> 5;                  // 0..5
  const int g = ntile * 16 + (lane & 15);
  const int bbase = blockIdx.x * 16;
  const int rowbase = (lane < 16) ? 0 : 8;

  // zero state
  for (int i = tid; i < 16 * 32; i += 192) { (&h0[0][0])[i] = (__bf16)0.0f; (&h1[0][0])[i] = (__bf16)0.0f; }
  for (int i = tid; i < 16 * HID; i += 192) { (&h0f[0][0])[i] = 0.0f; (&h1f[0][0])[i] = 0.0f; }
  __syncthreads();

  // invariant weight fragments + biases
  const v16bf Bhh0 = load_b_frag(whh0p, g, 32, 0, lane);
  const v16bf Bih1 = load_b_frag(wih1p, g, 32, 0, lane);
  const v16bf Bhh1 = load_b_frag(whh1p, g, 32, 0, lane);
  const float bHH0 = bhh0p[g];
  const float bIH1 = bih1p[g];
  const float bHH1 = bhh1p[g];

  const int bb = tid / HID;       // for gate phases (tid < 480)
  const int jj = tid % HID;

  for (int t = 0; t < TT; ++t) {
    // phase A: hp0 = h0 @ Whh0^T + bhh0
    {
      v16bf a = load_a_frag(&h0[0][0], 32, 0, lane);
      v8f acc;
#pragma unroll
      for (int r = 0; r < 8; ++r) acc[r] = bHH0;
      acc = wmma_bf16(a, Bhh0, acc);
#pragma unroll
      for (int r = 0; r < 8; ++r) hpA[rowbase + r][g] = acc[r];
    }
    __syncthreads();
    // phase B: layer-0 gates
    if (tid < 16 * HID) {
      size_t base = ((size_t)t * NB + bbase + bb) * GPAD;
      float r = 1.0f / (1.0f + expf(-(xp0[base + jj]            + hpA[bb][jj])));
      float z = 1.0f / (1.0f + expf(-(xp0[base + HID + jj]      + hpA[bb][HID + jj])));
      float n = tanhf(xp0[base + 2 * HID + jj] + r * hpA[bb][2 * HID + jj]);
      float hn = (1.0f - z) * n + z * h0f[bb][jj];
      h0f[bb][jj] = hn;
      h0[bb][jj] = (__bf16)hn;
    }
    __syncthreads();
    // phase C: xp1 = h0 @ Wih1^T + bih1 ; hp1 = h1 @ Whh1^T + bhh1
    {
      v16bf a0 = load_a_frag(&h0[0][0], 32, 0, lane);
      v8f acc0;
#pragma unroll
      for (int r = 0; r < 8; ++r) acc0[r] = bIH1;
      acc0 = wmma_bf16(a0, Bih1, acc0);
      v16bf a1 = load_a_frag(&h1[0][0], 32, 0, lane);
      v8f acc1;
#pragma unroll
      for (int r = 0; r < 8; ++r) acc1[r] = bHH1;
      acc1 = wmma_bf16(a1, Bhh1, acc1);
#pragma unroll
      for (int r = 0; r < 8; ++r) {
        hpA[rowbase + r][g] = acc0[r];
        hpB[rowbase + r][g] = acc1[r];
      }
    }
    __syncthreads();
    // phase D: layer-1 gates
    if (tid < 16 * HID) {
      float r = 1.0f / (1.0f + expf(-(hpA[bb][jj]           + hpB[bb][jj])));
      float z = 1.0f / (1.0f + expf(-(hpA[bb][HID + jj]     + hpB[bb][HID + jj])));
      float n = tanhf(hpA[bb][2 * HID + jj] + r * hpB[bb][2 * HID + jj]);
      float hn = (1.0f - z) * n + z * h1f[bb][jj];
      h1f[bb][jj] = hn;
      h1[bb][jj] = (__bf16)hn;
    }
    __syncthreads();
  }
  if (tid < 16 * HID) h1out[(size_t)(bbase + bb) * HID + jj] = h1f[bb][jj];
}

// ---------------- kernel 6: final global BN ----------------
__global__ __launch_bounds__(256) void k_outbn(const float* __restrict__ h1out,
                                               const float* __restrict__ bnp,
                                               float* __restrict__ out) {
  __shared__ float rs[256], rq[256];
  const int n = NB * HID;
  float s = 0.0f, q = 0.0f;
  for (int i = threadIdx.x; i < n; i += 256) {
    float v = h1out[i];
    s += v; q += v * v;
  }
  rs[threadIdx.x] = s; rq[threadIdx.x] = q;
  __syncthreads();
  for (int off = 128; off > 0; off >>= 1) {
    if (threadIdx.x < off) { rs[threadIdx.x] += rs[threadIdx.x + off]; rq[threadIdx.x] += rq[threadIdx.x + off]; }
    __syncthreads();
  }
  float mu = rs[0] / (float)n;
  float var = rq[0] / (float)n - mu * mu;
  float a = bnp[14] * rsqrtf(var + 1e-5f);
  float c = bnp[15] - mu * a;
  for (int i = threadIdx.x; i < n; i += 256) out[i] = h1out[i] * a + c;
}

// ---------------- launcher ----------------
extern "C" void kernel_launch(void* const* d_in, const int* in_sizes, int n_in,
                              void* d_out, int out_size, void* d_ws, size_t ws_size,
                              hipStream_t stream) {
  (void)in_sizes; (void)n_in; (void)out_size; (void)ws_size;
  const float* data1 = (const float*)d_in[0];
  const float* bnp   = (const float*)d_in[1];
  const float* Wih0  = (const float*)d_in[2];
  const float* Whh0  = (const float*)d_in[3];
  const float* bih0  = (const float*)d_in[4];
  const float* bhh0  = (const float*)d_in[5];
  const float* Wih1  = (const float*)d_in[6];
  const float* Whh1  = (const float*)d_in[7];
  const float* bih1  = (const float*)d_in[8];
  const float* bhh1  = (const float*)d_in[9];

  char* ws = (char*)d_ws;
  float*  stats = (float*)(ws + 0);          // 14 f32
  float*  ac    = (float*)(ws + 256);        // 14 f32 (a[7], c[7])
  float*  b0p   = (float*)(ws + 512);        // 96 f32
  float*  bhh0p = (float*)(ws + 896);
  float*  bih1p = (float*)(ws + 1280);
  float*  bhh1p = (float*)(ws + 1664);
  __bf16* whh0p = (__bf16*)(ws + 2048);      // 96*32 bf16
  __bf16* wih1p = (__bf16*)(ws + 8192);
  __bf16* whh1p = (__bf16*)(ws + 14336);
  __bf16* wp0   = (__bf16*)(ws + 20480);     // 96*384 bf16
  float*  h1out = (float*)(ws + 94208);      // 256*30 f32
  __bf16* feat  = (__bf16*)(ws + 131072);    // 61440*384 bf16 (47.2 MB)
  float*  xp0   = (float*)(ws + 47316992ULL);// 61440*96 f32  (23.6 MB)

  k_init<<<1, 32, 0, stream>>>(stats);
  k_features<<<TT, 256, 0, stream>>>(data1, feat, stats);
  k_finalize<<<1, 32, 0, stream>>>(stats, bnp, ac);
  k_wprep<<<1, 128, 0, stream>>>(Wih0, bih0, Whh0, bhh0, Wih1, bih1, Whh1, bhh1,
                                 ac, wp0, b0p, whh0p, bhh0p, wih1p, bih1p, whh1p, bhh1p);
  k_gemm<<<MROWS / 16, 192, 0, stream>>>(feat, wp0, b0p, xp0);
  k_scan<<<NB / 16, 192, 0, stream>>>(xp0, whh0p, bhh0p, wih1p, bih1p, whh1p, bhh1p, h1out);
  k_outbn<<<1, 256, 0, stream>>>(h1out, bnp, (float*)d_out);
}
